// TreeMeshGPT_79748952752497
// MI455X (gfx1250) — compile-verified, matrix operand order checked by reading, not run on hardware
//
#include <hip/hip_runtime.h>
#include <hip/hip_bf16.h>
#include <math.h>

// ---- problem constants (match reference) ----
#define D_    1024
#define H_    16
#define DH_   64
#define L_    24
#define S_    4096
#define SP1_  4097
#define DFF_  4096
#define NC_   128
#define TOPK_ 10
#define EOS_  129           // NC + 1
#define SCALE_ 0.125f       // 1/sqrt(64)
#define SPLIT_ 8            // split-K factor for big GEMVs
#define NCHUNK_ 8           // s-chunks for attention output

typedef __attribute__((ext_vector_type(16))) _Float16 v16h;
typedef __attribute__((ext_vector_type(8)))  float    v8f;
typedef __attribute__((ext_vector_type(4)))  unsigned v4u;
typedef __attribute__((ext_vector_type(8)))  int      v8i;
typedef __attribute__((ext_vector_type(4)))  int      v4i;

// ---------------------------------------------------------------------------
// LayerNorm: one block, 256 threads; writes f32 and (optionally) f16 copies.
// ---------------------------------------------------------------------------
__global__ void ln_kernel(const float* __restrict__ x, const float* __restrict__ g,
                          float* __restrict__ out, _Float16* __restrict__ outh, int n) {
    __shared__ float s1[256], s2[256];
    int tid = threadIdx.x;
    float a = 0.f, b = 0.f;
    for (int i = tid; i < n; i += 256) { float v = x[i]; a += v; b += v * v; }
    s1[tid] = a; s2[tid] = b; __syncthreads();
    for (int off = 128; off > 0; off >>= 1) {
        if (tid < off) { s1[tid] += s1[tid + off]; s2[tid] += s2[tid + off]; }
        __syncthreads();
    }
    float m = s1[0] / n;
    float var = s2[0] / n - m * m;
    float r = rsqrtf(var + 1e-5f);
    for (int i = tid; i < n; i += 256) {
        float v = (x[i] - m) * r * g[i];
        out[i] = v;
        if (outh) outh[i] = (_Float16)v;
    }
}

// ---------------------------------------------------------------------------
// Fused WMMA GEMV (per-lane gather) — used for small ragged heads (N=130/128).
// ---------------------------------------------------------------------------
template <int ACT>
__global__ void gemv_wmma_kernel(const _Float16* __restrict__ xh,
                                 const float* __restrict__ W,
                                 const float* __restrict__ bias,
                                 const float* __restrict__ add,
                                 float* __restrict__ out,
                                 _Float16* __restrict__ outh,
                                 int K, int N) {
    int wave = (blockIdx.x * blockDim.x + threadIdx.x) >> 5;
    int lane = threadIdx.x & 31;
    int n0 = wave * 16;
    if (n0 >= N) return;

    int half = (lane & 16) ? 8 : 0;
    int nl = lane & 15;
    int cn = n0 + nl; if (cn > N - 1) cn = N - 1;

    v8f c = {};
    for (int k0 = 0; k0 < K; k0 += 32) {
        v16h a;
        #pragma unroll
        for (int e = 0; e < 8; ++e) {
            a[e]     = xh[k0 + half + e];
            a[8 + e] = xh[k0 + 16 + half + e];
        }
        const float* Wc = W + (size_t)k0 * N + cn;
        __builtin_prefetch(Wc + (size_t)32 * N, 0, 1);
        v16h b;
        #pragma unroll
        for (int e = 0; e < 8; ++e) {
            b[e]     = (_Float16)Wc[(size_t)(half + e) * N];
            b[8 + e] = (_Float16)Wc[(size_t)(16 + half + e) * N];
        }
        c = __builtin_amdgcn_wmma_f32_16x16x32_f16(false, a, false, b,
                                                   (short)0, c, false, false);
    }
    if (lane < 16 && (n0 + lane) < N) {
        int n = n0 + lane;
        float r = c[0];
        if (bias) r += bias[n];
        if (ACT == 1) r = fmaxf(r, 0.f);
        if (add) r += add[n];
        out[n] = r;
        if (outh) outh[n] = (_Float16)r;
    }
}

// ---------------------------------------------------------------------------
// TDM + split-K WMMA GEMV. Block = 128 threads (4 waves) = 64 output columns.
// Wave 0 drives the Tensor Data Mover: 32x64 f32 W-tiles stream into a
// double-buffered LDS stage (tensor_load_to_lds, s_wait_tensorcnt), WMMA
// consumes from LDS. gridDim.y = split-K index; partials reduced later.
// ---------------------------------------------------------------------------
__global__ void gemv_wmma_tdm_kernel(const _Float16* __restrict__ xh,
                                     const float* __restrict__ W,
                                     float* __restrict__ part,
                                     int K, int N, int Kc) {
    __shared__ float tile[2][32 * 64];   // 2 x 8KB double buffer
    int wave = threadIdx.x >> 5;
    int lane = threadIdx.x & 31;
    int n0col = blockIdx.x * 64;
    int kbase = blockIdx.y * Kc;
    int half = (lane & 16) ? 8 : 0;
    int nl = lane & 15;
    int col = wave * 16 + nl;
    bool issuer = (wave == 0);

    // D# group1 (constant): data_size=4B, tensor = N x K, tile = 64 x 32, stride0 = N
    v8i g1;
    g1[0] = (2 << 16);                                   // data_size code 2 -> 4 bytes
    g1[1] = (int)(((unsigned)N & 0xFFFFu) << 16);        // tensor_dim0 [15:0]
    g1[2] = (int)((((unsigned)N >> 16) & 0xFFFFu) |
                  (((unsigned)K & 0xFFFFu) << 16));      // tensor_dim0 hi | tensor_dim1 lo
    g1[3] = (int)((((unsigned)K >> 16) & 0xFFFFu) |
                  (64u << 16));                          // tensor_dim1 hi | tile_dim0=64
    g1[4] = 32;                                          // tile_dim1=32, tile_dim2=0
    g1[5] = N;                                           // tensor_dim0_stride lo32
    g1[6] = 0;
    g1[7] = 0;
    v4i z4 = {0, 0, 0, 0};
    v8i z8 = {0, 0, 0, 0, 0, 0, 0, 0};

    auto issue = [&](int k0, int buf) {
        unsigned lds = (unsigned)(size_t)(&tile[buf][0]);
        unsigned long long ga =
            (unsigned long long)(const void*)(W + (size_t)(kbase + k0) * N + n0col);
        v4u g0;
        g0[0] = 1u;                                      // count=1, user mode
        g0[1] = lds;                                     // lds_addr
        g0[2] = (unsigned)ga;                            // global_addr lo
        g0[3] = (unsigned)(ga >> 32) | (2u << 30);       // global_addr hi | type=2
        __builtin_amdgcn_tensor_load_to_lds(g0, g1, z4, z4, z8, 0);
    };

    if (issuer) {
        issue(0, 0);
        if (Kc > 32) issue(32, 1);
    }

    v8f c = {};
    for (int k0 = 0; k0 < Kc; k0 += 32) {
        int buf = (k0 >> 5) & 1;
        if (issuer) {
            if (k0 + 32 < Kc) __builtin_amdgcn_s_wait_tensorcnt(1);
            else              __builtin_amdgcn_s_wait_tensorcnt(0);
        }
        __syncthreads();                 // tile[buf] now valid for all waves
        v16h a, b;
        #pragma unroll
        for (int e = 0; e < 8; ++e) {
            a[e]     = xh[kbase + k0 + half + e];
            a[8 + e] = xh[kbase + k0 + 16 + half + e];
            b[e]     = (_Float16)tile[buf][(half + e) * 64 + col];
            b[8 + e] = (_Float16)tile[buf][(16 + half + e) * 64 + col];
        }
        c = __builtin_amdgcn_wmma_f32_16x16x32_f16(false, a, false, b,
                                                   (short)0, c, false, false);
        __syncthreads();                 // buffer free to refill
        if (issuer && (k0 + 64 < Kc)) issue(k0 + 64, buf);
    }
    if (lane < 16)
        part[(size_t)blockIdx.y * N + n0col + wave * 16 + lane] = c[0];
}

// sum split-K partials, apply bias/act/residual, emit f32 + f16
template <int ACT>
__global__ void reduce_split_kernel(const float* __restrict__ part, int split,
                                    const float* __restrict__ bias,
                                    const float* __restrict__ add,
                                    float* __restrict__ out,
                                    _Float16* __restrict__ outh, int N) {
    int n = blockIdx.x * blockDim.x + threadIdx.x;
    if (n >= N) return;
    float r = 0.f;
    for (int s = 0; s < split; ++s) r += part[(size_t)s * N + n];
    if (bias) r += bias[n];
    if (ACT == 1) r = fmaxf(r, 0.f);
    if (add) r += add[n];
    out[n] = r;
    if (outh) outh[n] = (_Float16)r;
}

// ---------------------------------------------------------------------------
// Attention
// ---------------------------------------------------------------------------
__global__ void attn_scores_kernel(const float* __restrict__ q,
                                   const float* __restrict__ Kc,
                                   const float* __restrict__ kn,
                                   float* __restrict__ scores) {
    int h = blockIdx.y;
    int s = blockIdx.x * blockDim.x + threadIdx.x;
    if (s >= SP1_) return;
    const float4* qp = (const float4*)(q + h * DH_);
    const float4* kp = (s < S_) ? (const float4*)(Kc + ((size_t)s * H_ + h) * DH_)
                                : (const float4*)(kn + h * DH_);
    float acc = 0.f;
    #pragma unroll
    for (int i = 0; i < 16; ++i) {
        float4 a = qp[i]; float4 b = kp[i];
        acc += a.x * b.x + a.y * b.y + a.z * b.z + a.w * b.w;
    }
    scores[(size_t)h * SP1_ + s] = acc * SCALE_;
}

__global__ void attn_softmax_kernel(float* __restrict__ scores) {
    int h = blockIdx.x;
    float* sc = scores + (size_t)h * SP1_;
    __shared__ float red[256];
    int tid = threadIdx.x;
    float mx = -3.4e38f;
    for (int i = tid; i < SP1_; i += 256) mx = fmaxf(mx, sc[i]);
    red[tid] = mx; __syncthreads();
    for (int o = 128; o > 0; o >>= 1) { if (tid < o) red[tid] = fmaxf(red[tid], red[tid + o]); __syncthreads(); }
    mx = red[0]; __syncthreads();
    float sum = 0.f;
    for (int i = tid; i < SP1_; i += 256) { float e = __expf(sc[i] - mx); sc[i] = e; sum += e; }
    red[tid] = sum; __syncthreads();
    for (int o = 128; o > 0; o >>= 1) { if (tid < o) red[tid] += red[tid + o]; __syncthreads(); }
    float inv = 1.f / red[0];
    for (int i = tid; i < SP1_; i += 256) sc[i] *= inv;
}

// partial o over NCHUNK_ s-chunks: opart[(chunk*H + h)*DH + d]
__global__ void attn_out_kernel(const float* __restrict__ scores,
                                const float* __restrict__ Vc,
                                const float* __restrict__ vn,
                                float* __restrict__ opart) {
    int h = blockIdx.x;
    int chunk = blockIdx.y;
    int d = threadIdx.x & 63;
    int si = threadIdx.x >> 6;  // 0..3
    const int CL = S_ / NCHUNK_;  // 512
    int s0 = chunk * CL;
    const float* sc = scores + (size_t)h * SP1_;
    float acc = 0.f;
    for (int s = s0 + si; s < s0 + CL; s += 4)
        acc += sc[s] * Vc[((size_t)s * H_ + h) * DH_ + d];
    if (chunk == NCHUNK_ - 1 && si == 0) acc += sc[S_] * vn[h * DH_ + d];
    __shared__ float sm[256];
    sm[threadIdx.x] = acc; __syncthreads();
    if (si == 0) {
        float r = sm[d] + sm[64 + d] + sm[128 + d] + sm[192 + d];
        opart[((size_t)chunk * H_ + h) * DH_ + d] = r;
    }
}

__global__ void attn_out_reduce_kernel(const float* __restrict__ opart,
                                       _Float16* __restrict__ oh) {
    int i = blockIdx.x * blockDim.x + threadIdx.x;  // H*DH = 1024
    if (i >= H_ * DH_) return;
    float r = 0.f;
    #pragma unroll
    for (int c = 0; c < NCHUNK_; ++c) r += opart[(size_t)c * H_ * DH_ + i];
    oh[i] = (_Float16)r;
}

// ---------------------------------------------------------------------------
// gelu(g)*u -> f16 staging for Wd GEMV
__global__ void glu_kernel(const float* __restrict__ g, const float* __restrict__ u,
                           _Float16* __restrict__ outh, int n) {
    int i = blockIdx.x * blockDim.x + threadIdx.x;
    if (i >= n) return;
    float x = g[i];
    float t = tanhf(0.7978845608028654f * (x + 0.044715f * x * x * x));
    outh[i] = (_Float16)(0.5f * x * (1.f + t) * u[i]);
}

__global__ void gather_emb_kernel(const float* __restrict__ emb, const int* __restrict__ idx,
                                  int clampMax, _Float16* __restrict__ dst) {
    int i = blockIdx.x * blockDim.x + threadIdx.x;
    if (i >= D_) return;
    int r = idx[0]; if (r > clampMax) r = clampMax; if (r < 0) r = 0;
    dst[i] = (_Float16)emb[(size_t)r * D_ + i];
}

// softmax -> top-10 -> (EOS-masked) categorical sample, deterministic hash RNG.
__global__ void sample_kernel(const float* __restrict__ logits, int n, int eos,
                              unsigned seed, int* __restrict__ outIdx) {
    if (threadIdx.x != 0 || blockIdx.x != 0) return;
    float p[130];
    float mx = -3.4e38f;
    for (int i = 0; i < n; ++i) mx = fmaxf(mx, logits[i]);
    float sum = 0.f;
    for (int i = 0; i < n; ++i) { p[i] = __expf(logits[i] - mx); sum += p[i]; }
    float inv = 1.f / sum;
    for (int i = 0; i < n; ++i) p[i] *= inv;
    int ti[TOPK_]; float tp[TOPK_];
    for (int k = 0; k < TOPK_; ++k) {
        int bi = 0; float bv = -1.f;
        for (int i = 0; i < n; ++i) {
            bool taken = false;
            for (int j = 0; j < k; ++j) if (ti[j] == i) { taken = true; break; }
            if (!taken && p[i] > bv) { bv = p[i]; bi = i; }
        }
        ti[k] = bi; tp[k] = bv;
    }
    int result;
    if (eos >= 0 && ti[0] == eos) {
        result = eos;
    } else {
        float w[TOPK_]; float tot = 0.f;
        for (int k = 0; k < TOPK_; ++k) {
            w[k] = (eos >= 0 && ti[k] == eos) ? 0.f : tp[k];
            tot += w[k];
        }
        unsigned s = seed * 2654435761u;
        s ^= s >> 16; s *= 2246822519u; s ^= s >> 13; s *= 3266489917u; s ^= s >> 16;
        float u = ((float)(s & 0xFFFFFFu) / 16777216.f) * tot;
        int pick = 0; float c = 0.f;
        for (int k = 0; k < TOPK_; ++k) { c += w[k]; pick = k; if (u <= c) break; }
        result = ti[pick];
    }
    outIdx[0] = result;
}

__global__ void finalize_kernel(const float* __restrict__ lg,
                                const int* __restrict__ xi, const int* __restrict__ yi,
                                const int* __restrict__ zi, float* __restrict__ out) {
    int i = blockIdx.x * blockDim.x + threadIdx.x;
    if (i == 0) out[0] = (float)xi[0];
    if (i == 1) out[1] = (float)yi[0];
    if (i == 2) out[2] = (float)zi[0];
    if (i < 386) out[3 + i] = lg[i];
}

// ---------------------------------------------------------------------------
extern "C" void kernel_launch(void* const* d_in, const int* in_sizes, int n_in,
                              void* d_out, int out_size, void* d_ws, size_t ws_size,
                              hipStream_t stream) {
    (void)in_sizes; (void)n_in; (void)out_size; (void)ws_size;

    const float* x_in   = (const float*)d_in[0];
    const float* kcache = (const float*)d_in[1];
    const float* vcache = (const float*)d_in[2];
    const float* ln1    = (const float*)d_in[3];
    const float* Wq     = (const float*)d_in[4];
    const float* Wk     = (const float*)d_in[5];
    const float* Wv     = (const float*)d_in[6];
    const float* Wo     = (const float*)d_in[7];
    const float* ln2    = (const float*)d_in[8];
    const float* Wg     = (const float*)d_in[9];
    const float* Wu     = (const float*)d_in[10];
    const float* Wd     = (const float*)d_in[11];
    const float* lnf    = (const float*)d_in[12];
    const float* h1w1 = (const float*)d_in[13]; const float* h1b1 = (const float*)d_in[14];
    const float* h1w2 = (const float*)d_in[15]; const float* h1b2 = (const float*)d_in[16];
    const float* h1w3 = (const float*)d_in[17]; const float* h1b3 = (const float*)d_in[18];
    const float* h2w1 = (const float*)d_in[19]; const float* h2b1 = (const float*)d_in[20];
    const float* h2w2 = (const float*)d_in[21]; const float* h2b2 = (const float*)d_in[22];
    const float* h2w3 = (const float*)d_in[23]; const float* h2b3 = (const float*)d_in[24];
    const float* h3w1 = (const float*)d_in[25]; const float* h3b1 = (const float*)d_in[26];
    const float* h3w2 = (const float*)d_in[27]; const float* h3b2 = (const float*)d_in[28];
    const float* h3w3 = (const float*)d_in[29]; const float* h3b3 = (const float*)d_in[30];
    const float* c1e  = (const float*)d_in[31];
    const float* c2e  = (const float*)d_in[32];

    // workspace layout (bytes)
    char* B = (char*)d_ws;
    float* h    = (float*)(B + 0);
    float* hn   = (float*)(B + 4096);
    float* qb   = (float*)(B + 8192);
    float* kb   = (float*)(B + 12288);
    float* vb   = (float*)(B + 16384);
    float* gb   = (float*)(B + 20480);      // 16KB
    float* ub   = (float*)(B + 36864);      // 16KB
    float* sc   = (float*)(B + 53248);      // 16*4097*4 = 262208 B
    float* resf = (float*)(B + 315648);
    float* t1   = (float*)(B + 319744);
    float* t2   = (float*)(B + 323840);
    float* lg   = (float*)(B + 327936);     // 386 floats: z[0:130] y[130:258] x[258:386]
    _Float16* hnh   = (_Float16*)(B + 329728);
    _Float16* oh_   = (_Float16*)(B + 331776);
    _Float16* ffh   = (_Float16*)(B + 333824);
    _Float16* cat2h = (_Float16*)(B + 342016);
    _Float16* cat3h = (_Float16*)(B + 346112);
    _Float16* t1h   = (_Float16*)(B + 352256);
    _Float16* t2h   = (_Float16*)(B + 354304);
    int* zidx = (int*)(B + 356352);
    int* yidx = (int*)(B + 356356);
    int* xidx = (int*)(B + 356360);
    float* part  = (float*)(B + 360448);    // 8*4096 floats = 128KB (split-K partials)
    float* opart = part;                    // aliased: attn partials (32KB), disjoint in time

    // big GEMV: TDM + split-K, N % 64 == 0, K % (SPLIT_*32) == 0
    auto gemv_big = [&](int act, const _Float16* xh, const float* W, const float* bias,
                        const float* add, float* out, _Float16* outh, int K, int N) {
        int Kc = K / SPLIT_;
        dim3 grid(N / 64, SPLIT_);
        gemv_wmma_tdm_kernel<<<grid, 128, 0, stream>>>(xh, W, part, K, N, Kc);
        int rb = (N + 255) / 256;
        if (act)
            reduce_split_kernel<1><<<rb, 256, 0, stream>>>(part, SPLIT_, bias, add, out, outh, N);
        else
            reduce_split_kernel<0><<<rb, 256, 0, stream>>>(part, SPLIT_, bias, add, out, outh, N);
    };
    // small ragged GEMV (fused)
    auto gemv_small = [&](int act, const _Float16* xh, const float* W, const float* bias,
                          const float* add, float* out, _Float16* outh, int K, int N) {
        int nw = (N + 15) / 16;
        int blocks = (nw + 3) / 4;
        if (act)
            gemv_wmma_kernel<1><<<blocks, 128, 0, stream>>>(xh, W, bias, add, out, outh, K, N);
        else
            gemv_wmma_kernel<0><<<blocks, 128, 0, stream>>>(xh, W, bias, add, out, outh, K, N);
    };

    (void)hipMemcpyAsync(h, x_in, D_ * sizeof(float), hipMemcpyDeviceToDevice, stream);

    const size_t DD = (size_t)D_ * D_;
    const size_t DF = (size_t)D_ * DFF_;
    const size_t KV = (size_t)S_ * H_ * DH_;

    for (int l = 0; l < L_; ++l) {
        // attention block
        ln_kernel<<<1, 256, 0, stream>>>(h, ln1 + (size_t)l * D_, hn, hnh, D_);
        gemv_big(0, hnh, Wq + l * DD, nullptr, nullptr, qb, nullptr, D_, D_);
        gemv_big(0, hnh, Wk + l * DD, nullptr, nullptr, kb, nullptr, D_, D_);
        gemv_big(0, hnh, Wv + l * DD, nullptr, nullptr, vb, nullptr, D_, D_);
        attn_scores_kernel<<<dim3((SP1_ + 255) / 256, H_), 256, 0, stream>>>(
            qb, kcache + (size_t)l * KV, kb, sc);
        attn_softmax_kernel<<<H_, 256, 0, stream>>>(sc);
        attn_out_kernel<<<dim3(H_, NCHUNK_), 256, 0, stream>>>(
            sc, vcache + (size_t)l * KV, vb, opart);
        attn_out_reduce_kernel<<<(H_ * DH_ + 255) / 256, 256, 0, stream>>>(opart, oh_);
        gemv_big(0, oh_, Wo + l * DD, nullptr, /*add=*/h, h, nullptr, D_, D_);

        // GLU FF block
        ln_kernel<<<1, 256, 0, stream>>>(h, ln2 + (size_t)l * D_, hn, hnh, D_);
        gemv_big(0, hnh, Wg + (size_t)l * DF, nullptr, nullptr, gb, nullptr, D_, DFF_);
        gemv_big(0, hnh, Wu + (size_t)l * DF, nullptr, nullptr, ub, nullptr, D_, DFF_);
        glu_kernel<<<(DFF_ + 255) / 256, 256, 0, stream>>>(gb, ub, ffh, DFF_);
        gemv_big(0, ffh, Wd + (size_t)l * DF, nullptr, /*add=*/h, h, nullptr, DFF_, D_);
    }

    // final LN: f32 -> resf, f16 -> cat2h[0:1024]; replicate into cat3h[0:1024]
    ln_kernel<<<1, 256, 0, stream>>>(h, lnf, resf, cat2h, D_);
    (void)hipMemcpyAsync(cat3h, cat2h, D_ * sizeof(_Float16), hipMemcpyDeviceToDevice, stream);

    // head 1: z
    gemv_big(1, cat2h, h1w1, h1b1, nullptr, t1, t1h, D_, D_);
    gemv_big(1, t1h,  h1w2, h1b2, nullptr, t2, t2h, D_, D_);
    gemv_small(0, t2h, h1w3, h1b3, nullptr, lg, nullptr, D_, NC_ + 2);
    sample_kernel<<<1, 1, 0, stream>>>(lg, NC_ + 2, EOS_, 1u, zidx);
    gather_emb_kernel<<<4, 256, 0, stream>>>(c1e, zidx, NC_ - 1, cat2h + D_);
    gather_emb_kernel<<<4, 256, 0, stream>>>(c1e, zidx, NC_ - 1, cat3h + D_);

    // head 2: y
    gemv_big(1, cat2h, h2w1, h2b1, nullptr, t1, t1h, 2 * D_, D_);
    gemv_big(1, t1h,  h2w2, h2b2, nullptr, t2, t2h, D_, D_);
    gemv_small(0, t2h, h2w3, h2b3, nullptr, lg + 130, nullptr, D_, NC_);
    sample_kernel<<<1, 1, 0, stream>>>(lg + 130, NC_, -1, 2u, yidx);
    gather_emb_kernel<<<4, 256, 0, stream>>>(c2e, yidx, NC_ - 1, cat3h + 2 * D_);

    // head 3: x
    gemv_big(1, cat3h, h3w1, h3b1, nullptr, t1, t1h, 3 * D_, D_);
    gemv_big(1, t1h,  h3w2, h3b2, nullptr, t2, t2h, D_, D_);
    gemv_small(0, t2h, h3w3, h3b3, nullptr, lg + 258, nullptr, D_, NC_);
    sample_kernel<<<1, 1, 0, stream>>>(lg + 258, NC_, -1, 3u, xidx);

    finalize_kernel<<<1, 512, 0, stream>>>(lg, xidx, yidx, zidx, (float*)d_out);
}
